// GCN_RNI_78683800863475
// MI455X (gfx1250) — matrix operand order repaired and checked by reading, not compiled, not run on hardware
//
#include <hip/hip_runtime.h>
#include <hip/hip_bf16.h>
#include <math.h>

// ---------------------------------------------------------------------------
// GCN + RNI for MI455X (gfx1250, wave32).
// Activations h are stored as bf16 (they feed only the GEMMs); GEMM output hw
// and edge-aggregation stay fp32. GEMMs: v_wmma_f32_16x16x32_bf16, one wave
// computes a 16 x (16*NT) tile with the K loop fully unrolled (CIN template).
// Edge aggregation = coalesced float4 gather + global_atomic_add_f32 scatter
// (per-layer hw[] is 51MB -> resident in the 192MB L2).
// ---------------------------------------------------------------------------

typedef __attribute__((ext_vector_type(16))) __bf16 v16bf;
typedef __attribute__((ext_vector_type(8)))  __bf16 v8bf;
typedef __attribute__((ext_vector_type(8)))  float  v8f;

__device__ __forceinline__ __bf16 f2bf(float f) {
  // round-to-nearest-even f32 -> bf16 via integer rounding
  unsigned u = __builtin_bit_cast(unsigned, f);
  u += 0x7FFFu + ((u >> 16) & 1u);
  unsigned short h = (unsigned short)(u >> 16);
  return __builtin_bit_cast(__bf16, h);
}

// ---------------- elementwise helpers ----------------

__global__ void fill_f32(float* __restrict__ p, float v, int n) {
  int i = blockIdx.x * blockDim.x + threadIdx.x;
  if (i < n) p[i] = v;
}

__global__ void deg_count(const int* __restrict__ dst, float* __restrict__ deg, int E) {
  int i = blockIdx.x * blockDim.x + threadIdx.x;
  if (i < E) atomicAdd(&deg[dst[i]], 1.0f);
}

__global__ void make_dinv(float* __restrict__ d, int n) {
  int i = blockIdx.x * blockDim.x + threadIdx.x;
  if (i < n) d[i] = rsqrtf(d[i] + 1.0f);   // deg includes self-loop
}

__global__ void convert_bf16(const float* __restrict__ src, __bf16* __restrict__ dst, int n) {
  int i = blockIdx.x * blockDim.x + threadIdx.x;
  if (i < n) dst[i] = f2bf(src[i]);
}

// Wt[c*Cin + k] = bf16( W[k*Cout + c] )   (transpose + narrow, tiny)
__global__ void pack_wt(const float* __restrict__ W, __bf16* __restrict__ Wt,
                        int Cin, int Cout) {
  int idx = blockIdx.x * blockDim.x + threadIdx.x;
  if (idx >= Cin * Cout) return;
  int c = idx / Cin;
  int k = idx - c * Cin;
  Wt[idx] = f2bf(W[(size_t)k * Cout + c]);
}

// ---------------- WMMA GEMM: C[N,Cout] = A[N,CIN] @ W ----------------
// One wave computes a 16 x (16*NT) output tile; K fully unrolled.
template <int CIN, int NT>
__global__ __launch_bounds__(256) void gemm_bf16_wmma(
    const __bf16* __restrict__ A, const __bf16* __restrict__ Wt,
    float* __restrict__ C, int N, int Cout) {
  const int lane      = threadIdx.x & 31;
  const int waveId    = blockIdx.x * 8 + (threadIdx.x >> 5);
  const int colGroups = Cout / (16 * NT);
  const int total     = (N >> 4) * colGroups;
  if (waveId >= total) return;                // wave-uniform: EXEC stays all-1s

  const int tileRow = waveId / colGroups;
  const int colBase = (waveId - tileRow * colGroups) * (16 * NT);
  const int half    = lane >> 4;              // 0: lanes 0-15, 1: lanes 16-31
  const int l16     = lane & 15;

  // A fragment: lane-half selects K base 0/16;
  // elements 0..7 -> K = kb..kb+7, elements 8..15 -> K = kb+8..kb+15.
  const __bf16* arow = A + (size_t)((tileRow << 4) + l16) * CIN + (half << 4);
  // B fragment: lanes 0-15 hold K = k0..k0+15, lanes 16-31 hold K = k0+16..k0+31
  // (Wt pre-transposed: contiguous 32B per fragment per lane).
  const __bf16* bptr[NT];
#pragma unroll
  for (int t = 0; t < NT; ++t)
    bptr[t] = Wt + (size_t)(colBase + t * 16 + l16) * CIN + (half << 4);

  v8f acc[NT];
#pragma unroll
  for (int t = 0; t < NT; ++t) acc[t] = (v8f){0.f, 0.f, 0.f, 0.f, 0.f, 0.f, 0.f, 0.f};

#pragma unroll
  for (int k0 = 0; k0 < CIN; k0 += 32) {
    v8bf alo = *(const v8bf*)(arow + k0);
    v8bf ahi = *(const v8bf*)(arow + k0 + 8);
    v16bf a = __builtin_shufflevector(alo, ahi, 0, 1, 2, 3, 4, 5, 6, 7,
                                      8, 9, 10, 11, 12, 13, 14, 15);
#pragma unroll
    for (int t = 0; t < NT; ++t) {
      v16bf b = *(const v16bf*)(bptr[t] + k0);
      acc[t] = __builtin_amdgcn_wmma_f32_16x16x32_bf16(
          /*neg_a=*/false, a, /*neg_b=*/false, b,
          /*c_mod=*/(short)0, acc[t], /*reuse_a=*/false, /*reuse_b=*/false);
    }
  }
  // C/D layout: VGPR r -> row tileRow*16 + half*8 + r, col = tile col + l16
  const size_t rowOff = (size_t)((tileRow << 4) + (half << 3)) * Cout;
#pragma unroll
  for (int t = 0; t < NT; ++t) {
    float* crow = C + rowOff + colBase + t * 16 + l16;
#pragma unroll
    for (int r = 0; r < 8; ++r) crow[(size_t)r * Cout] = acc[t][r];
  }
}

// ---------------- aggregation ----------------

// agg[i][c] = hw[i][c] * dinv[i]^2   (self-loop term)
__global__ void self_init(const float* __restrict__ hw, const float* __restrict__ dinv,
                          float* __restrict__ agg, int total, int cshift) {
  int i = blockIdx.x * blockDim.x + threadIdx.x;
  if (i >= total) return;
  float d = dinv[i >> cshift];
  agg[i] = hw[i] * d * d;
}

// agg[dst] += hw[src] * dinv[src]*dinv[dst]   (4 channels per thread)
__global__ void edge_scatter(const float* __restrict__ hw, const int* __restrict__ src,
                             const int* __restrict__ dst, const float* __restrict__ dinv,
                             float* __restrict__ agg, int E, int C, int vshift) {
  int idx = blockIdx.x * blockDim.x + threadIdx.x;
  int total = E << vshift;                     // E * C/4
  if (idx >= total) return;
  int e = idx >> vshift;
  int v = (idx & ((1 << vshift) - 1)) << 2;
  int s = src[e], d = dst[e];
  float nrm = dinv[s] * dinv[d];
  float4 h = *(const float4*)(hw + (size_t)s * C + v);
  float* out = agg + (size_t)d * C + v;
  atomicAdd(out + 0, h.x * nrm);
  atomicAdd(out + 1, h.y * nrm);
  atomicAdd(out + 2, h.z * nrm);
  atomicAdd(out + 3, h.w * nrm);
}

// h = elu(agg + b)  -> bf16 (GEMM-input layout)
__global__ void finalize_elu_bf16(const float* __restrict__ agg, const float* __restrict__ b,
                                  __bf16* __restrict__ h, int total, int cmask) {
  int i = blockIdx.x * blockDim.x + threadIdx.x;
  if (i >= total) return;
  float v = agg[i] + b[i & cmask];
  v = v > 0.f ? v : expm1f(v);
  h[i] = f2bf(v);
}

// layer-4 tail: out = agg + b (fp32, in place, no elu)
__global__ void finalize_bias_f32(float* __restrict__ x, const float* __restrict__ b,
                                  int total, int cmask) {
  int i = blockIdx.x * blockDim.x + threadIdx.x;
  if (i >= total) return;
  x[i] = x[i] + b[i & cmask];
}

// layer-2: H[:,0:32] = elu(agg2 + b2), H[:,32:256] = random_dims  (fused concat, bf16)
__global__ void finalize_l2_concat(const float* __restrict__ agg2, const float* __restrict__ b2,
                                   const float* __restrict__ rnd, __bf16* __restrict__ H, int N) {
  int i = blockIdx.x * blockDim.x + threadIdx.x;
  if (i >= N * 256) return;
  int node = i >> 8, c = i & 255;
  float v;
  if (c < 32) {
    v = agg2[(node << 5) + c] + b2[c];
    v = v > 0.f ? v : expm1f(v);
  } else {
    v = rnd[(size_t)node * 224 + (c - 32)];
  }
  H[i] = f2bf(v);
}

// ---------------------------------------------------------------------------

extern "C" void kernel_launch(void* const* d_in, const int* in_sizes, int n_in,
                              void* d_out, int out_size, void* d_ws, size_t ws_size,
                              hipStream_t stream) {
  const float* x    = (const float*)d_in[0];
  const int*   esrc = (const int*)d_in[1];
  const int*   edst = (const int*)d_in[2];
  const float* rnd  = (const float*)d_in[3];
  const float* W1   = (const float*)d_in[4];
  const float* b1   = (const float*)d_in[5];
  const float* W2   = (const float*)d_in[6];
  const float* b2   = (const float*)d_in[7];
  const float* W3   = (const float*)d_in[8];
  const float* b3   = (const float*)d_in[9];
  const float* W4   = (const float*)d_in[10];
  const float* b4   = (const float*)d_in[11];

  const int N = in_sizes[0] / 128;   // 50000
  const int E = in_sizes[1];         // 800000

  // workspace carve (256B aligned)
  char* ws = (char*)d_ws;
  size_t off = 0;
  auto carve = [&](size_t bytes) -> void* {
    void* p = ws + off;
    off += (bytes + 255) & ~(size_t)255;
    return p;
  };
  float*  dinv = (float*)carve((size_t)N * 4);
  float*  hw   = (float*)carve((size_t)N * 256 * 4);   // GEMM output (fp32)
  float*  agg  = (float*)carve((size_t)N * 256 * 4);   // aggregation buffer (fp32)
  __bf16* hbf  = (__bf16*)carve((size_t)N * 256 * 2);  // activations (bf16, GEMM input)
  __bf16* xbf  = (__bf16*)carve((size_t)N * 128 * 2);  // x in bf16
  __bf16* W1t  = (__bf16*)carve((size_t)128 * 256 * 2);
  __bf16* W2t  = (__bf16*)carve((size_t)256 * 32  * 2);
  __bf16* W3t  = (__bf16*)carve((size_t)256 * 256 * 2);
  __bf16* W4t  = (__bf16*)carve((size_t)256 * 256 * 2);
  (void)ws_size; (void)n_in; (void)out_size;

  const int TB = 256;
  auto blocks = [&](long long n) { return (int)((n + TB - 1) / TB); };

  // degrees -> dinv ; input conversion ; weight packing
  fill_f32    <<<blocks(N), TB, 0, stream>>>(dinv, 0.f, N);
  deg_count   <<<blocks(E), TB, 0, stream>>>(edst, dinv, E);
  make_dinv   <<<blocks(N), TB, 0, stream>>>(dinv, N);
  convert_bf16<<<blocks((long long)N * 128), TB, 0, stream>>>(x, xbf, N * 128);
  pack_wt<<<blocks(128 * 256), TB, 0, stream>>>(W1, W1t, 128, 256);
  pack_wt<<<blocks(256 * 32),  TB, 0, stream>>>(W2, W2t, 256, 32);
  pack_wt<<<blocks(256 * 256), TB, 0, stream>>>(W3, W3t, 256, 256);
  pack_wt<<<blocks(256 * 256), TB, 0, stream>>>(W4, W4t, 256, 256);

  auto gemmGrid = [&](int Cout, int NT) { return (((N >> 4) * (Cout / (16 * NT))) + 7) / 8; };

  // Layer 1: xbf[N,128] @ W1 -> hw[N,256] ; aggregate -> agg ; elu -> hbf
  gemm_bf16_wmma<128, 4><<<gemmGrid(256, 4), 256, 0, stream>>>(xbf, W1t, hw, N, 256);
  self_init   <<<blocks((long long)N * 256), TB, 0, stream>>>(hw, dinv, agg, N * 256, 8);
  edge_scatter<<<blocks((long long)E << 6), TB, 0, stream>>>(hw, esrc, edst, dinv, agg, E, 256, 6);
  finalize_elu_bf16<<<blocks((long long)N * 256), TB, 0, stream>>>(agg, b1, hbf, N * 256, 255);

  // Layer 2: hbf[N,256] @ W2 -> hw2[N,32] ; agg2 ; fused concat -> hbf
  float* hw2  = hw;
  float* agg2 = hw + (size_t)N * 32;
  gemm_bf16_wmma<256, 2><<<gemmGrid(32, 2), 256, 0, stream>>>(hbf, W2t, hw2, N, 32);
  self_init   <<<blocks((long long)N * 32), TB, 0, stream>>>(hw2, dinv, agg2, N * 32, 5);
  edge_scatter<<<blocks((long long)E << 3), TB, 0, stream>>>(hw2, esrc, edst, dinv, agg2, E, 32, 3);
  finalize_l2_concat<<<blocks((long long)N * 256), TB, 0, stream>>>(agg2, b2, rnd, hbf, N);

  // Layer 3: hbf @ W3 -> hw ; aggregate -> agg ; elu -> hbf
  gemm_bf16_wmma<256, 4><<<gemmGrid(256, 4), 256, 0, stream>>>(hbf, W3t, hw, N, 256);
  self_init   <<<blocks((long long)N * 256), TB, 0, stream>>>(hw, dinv, agg, N * 256, 8);
  edge_scatter<<<blocks((long long)E << 6), TB, 0, stream>>>(hw, esrc, edst, dinv, agg, E, 256, 6);
  finalize_elu_bf16<<<blocks((long long)N * 256), TB, 0, stream>>>(agg, b3, hbf, N * 256, 255);

  // Layer 4: hbf @ W4 -> hw ; aggregate directly into d_out ; + b4 (no elu)
  float* out = (float*)d_out;
  gemm_bf16_wmma<256, 4><<<gemmGrid(256, 4), 256, 0, stream>>>(hbf, W4t, hw, N, 256);
  self_init   <<<blocks((long long)N * 256), TB, 0, stream>>>(hw, dinv, out, N * 256, 8);
  edge_scatter<<<blocks((long long)E << 6), TB, 0, stream>>>(hw, esrc, edst, dinv, out, E, 256, 6);
  finalize_bias_f32<<<blocks((long long)N * 256), TB, 0, stream>>>(out, b4, N * 256, 255);
}
